// PointEncoder_5781025980874
// MI455X (gfx1250) — matrix operand order repaired and checked by reading, not compiled, not run on hardware
//
#include <hip/hip_runtime.h>
#include <math.h>

// ---------------- problem constants ----------------
#define MD     384          // model dim D
#define BATCH  16
#define NPTS   8192
#define LQ     512
#define FFI    1536         // FF inner dim (per GEGLU half)
#define TOTROWS (BATCH*NPTS)

// ---------------- WMMA types / helpers ----------------
typedef __attribute__((ext_vector_type(16))) __bf16 v16bf;
typedef __attribute__((ext_vector_type(8)))  float  v8f;
typedef unsigned u32x4 __attribute__((ext_vector_type(4)));

union FragU { u32x4 q[2]; v16bf v; };

__device__ __forceinline__ unsigned short f2bf(float f) {
  union { float f; unsigned u; } c; c.f = f;
  unsigned u = c.u;
  u += 0x7FFFu + ((u >> 16) & 1u);          // round-to-nearest-even
  return (unsigned short)(u >> 16);
}
__device__ __forceinline__ float bf2f(unsigned short h) {
  union { unsigned u; float f; } c; c.u = ((unsigned)h) << 16; return c.f;
}

__device__ __forceinline__ v8f wmma_bf16(v16bf a, v16bf b, v8f c) {
  // v_wmma_f32_16x16x32_bf16  D = A(16x32) x B(32x16) + C
  return __builtin_amdgcn_wmma_f32_16x16x32_bf16(
      false, a, false, b, (short)0, c, false, false);
}

// 32 contiguous bytes -> one fragment (16 bf16)
__device__ __forceinline__ v16bf load_frag32B(const unsigned short* p) {
  FragU u;
  const u32x4* q = (const u32x4*)p;
  u.q[0] = q[0]; u.q[1] = q[1];
  return u.v;
}

// A-fragment from a row-major bf16 buffer (LDS or global).
// Contiguous k order: lane half h holds K = kt*32 + h*16 + (0..15).
__device__ __forceinline__ v16bf load_a_frag(const unsigned short* buf,
                                             int rowBase, int stride, int kt, int lane) {
  int m = rowBase + (lane & 15);
  int h = (lane >> 4) * 16;
  const unsigned short* base = buf + (size_t)m * stride + kt * 32 + h;
  FragU u;
  u.q[0] = *(const u32x4*)(base);
  u.q[1] = *(const u32x4*)(base + 8);
  return u.v;
}

// B-fragment from pre-swizzled weights: tile (ct,kt) is 512 contiguous bf16,
// lane l owns elements [l*16 .. l*16+15].
__device__ __forceinline__ v16bf load_b_frag_swz(const unsigned short* W,
                                                 int ct, int kt, int KT, int lane) {
  return load_frag32B(W + ((size_t)(ct * KT + kt)) * 512 + lane * 16);
}

// B-fragment where the K dimension is contiguous in memory per B-column:
// lane l -> column rowIdx(l), 16 contiguous bf16 starting at kbase(l).
__device__ __forceinline__ v16bf load_b_frag_rm(const unsigned short* base,
                                                size_t rowIdx, size_t stride, size_t kbase) {
  return load_frag32B(base + rowIdx * stride + kbase);
}

// K=384 GEMM micro-kernel, two B tiles in flight so global loads overlap the
// matrix pipe (partial s_wait_loadcnt instead of full stalls).
__device__ __forceinline__ v8f gemm_k12(const v16bf* afr,
                                        const unsigned short* __restrict__ W,
                                        int ct, int lane, v8f acc) {
  v16bf b0 = load_b_frag_swz(W, ct, 0, 12, lane);
  v16bf b1 = load_b_frag_swz(W, ct, 1, 12, lane);
#pragma unroll
  for (int kt = 0; kt < 12; kt += 2) {
    v16bf n0 = b0, n1 = b1;
    if (kt + 2 < 12) n0 = load_b_frag_swz(W, ct, kt + 2, 12, lane);
    if (kt + 3 < 12) n1 = load_b_frag_swz(W, ct, kt + 3, 12, lane);
    acc = wmma_bf16(afr[kt],     b0, acc);
    acc = wmma_bf16(afr[kt + 1], b1, acc);
    b0 = n0; b1 = n1;
  }
  return acc;
}

// ---------------- weight pre-swizzle (fp32 -> bf16 B-fragment layout) ----------------
__global__ void swz_kernel(const float* __restrict__ src, unsigned short* __restrict__ dst,
                           int K, int N) {
  int KT = K >> 5;
  size_t total = (size_t)K * N;
  for (size_t idx = (size_t)blockIdx.x * blockDim.x + threadIdx.x; idx < total;
       idx += (size_t)gridDim.x * blockDim.x) {
    int tile   = (int)(idx >> 9);
    int within = (int)(idx & 511);
    int lane = within >> 4;
    int j    = within & 15;
    int ct = tile / KT;
    int kt = tile - ct * KT;
    int k = kt * 32 + (lane >> 4) * 16 + j;
    int n = ct * 16 + (lane & 15);
    dst[idx] = f2bf(src[(size_t)k * N + n]);
  }
}

// One dense D->D layer from LDS(cur) to LDS(nxt).  Weight pointer comes
// directly from a kernarg so the B loads lower to global_load_b128.
template <bool RELU>
__device__ __forceinline__ void mlp_layer(const unsigned short* cur,
                                          unsigned short* nxt,
                                          const unsigned short* __restrict__ Wswz,
                                          const float* __restrict__ bias,
                                          int wave, int lane) {
  int h = lane >> 4, nn = lane & 15;
  v16bf afr[12];
#pragma unroll
  for (int kt = 0; kt < 12; ++kt) afr[kt] = load_a_frag(cur, wave * 16, MD, kt, lane);
#pragma unroll 1
  for (int ct = 0; ct < 24; ++ct) {
    v8f acc = {};
    acc = gemm_k12(afr, Wswz, ct, lane, acc);
    int n = ct * 16 + nn;
    float bv = bias[n];
#pragma unroll
    for (int r = 0; r < 8; ++r) {
      float v = acc[r] + bv;
      if (RELU) v = fmaxf(v, 0.f);
      nxt[(wave * 16 + h * 8 + r) * MD + n] = f2bf(v);
    }
  }
}

// ---------------- fused point MLP + LN(ctx) + kv projection ----------------
// grid = TOTROWS/128 blocks x 256 threads.  Emits k row-major, v transposed.
__global__ void __launch_bounds__(256, 1)
mlp_kernel(const float* __restrict__ x,
           const float* __restrict__ w0, const float* __restrict__ b0,
           const unsigned short* __restrict__ w1s, const float* __restrict__ b1,
           const unsigned short* __restrict__ w2s, const float* __restrict__ b2,
           const unsigned short* __restrict__ w3s, const float* __restrict__ b3,
           const float* __restrict__ lng, const float* __restrict__ lnb,
           const unsigned short* __restrict__ wkvs,
           unsigned short* __restrict__ kout,
           unsigned short* __restrict__ vTout) {
  extern __shared__ char smem[];
  unsigned short* bufA = (unsigned short*)smem;          // [128][384] bf16
  unsigned short* bufB = bufA + 128 * MD;                // [128][384] bf16
  float* stats = (float*)(bufB + 128 * MD);              // [128][2]
  int t = threadIdx.x, lane = t & 31, wave = t >> 5;
  int h = lane >> 4, nn = lane & 15;
  size_t r0 = (size_t)blockIdx.x * 128;

  // stage 0: h0 = relu(x @ w0 + b0)   (3 -> 384, pure VALU)
  for (int e = t; e < 128 * MD; e += 256) {
    int rr = e / MD, d = e - rr * MD;
    size_t row = r0 + rr;
    float acc = b0[d] + x[row * 3 + 0] * w0[0 * MD + d]
                      + x[row * 3 + 1] * w0[1 * MD + d]
                      + x[row * 3 + 2] * w0[2 * MD + d];
    bufA[e] = f2bf(fmaxf(acc, 0.f));
  }
  __syncthreads();

  // three dense layers, explicitly unrolled (keeps global addrspace inference)
  mlp_layer<true >(bufA, bufB, w1s, b1, wave, lane);
  __syncthreads();
  mlp_layer<true >(bufB, bufA, w2s, b2, wave, lane);
  __syncthreads();
  mlp_layer<false>(bufA, bufB, w3s, b3, wave, lane);
  __syncthreads();

  // LayerNorm(ctx) per row; ctx lives in bufB
  if (t < 128) {
    float s = 0.f, ss = 0.f;
#pragma unroll 1
    for (int d = 0; d < MD; ++d) { float v = bf2f(bufB[t * MD + d]); s += v; ss += v * v; }
    float mean = s / MD;
    float var  = ss / MD - mean * mean;
    stats[2 * t] = mean;
    stats[2 * t + 1] = rsqrtf(var + 1e-5f);
  }
  __syncthreads();
  for (int e = t; e < 128 * MD; e += 256) {
    int rr = e / MD, d = e - rr * MD;
    float v = bf2f(bufB[e]);
    bufA[e] = f2bf((v - stats[2 * rr]) * stats[2 * rr + 1] * lng[d] + lnb[d]);
  }
  __syncthreads();

  // kv = cn @ wkv : ct 0..23 -> k (row-major), ct 24..47 -> v (transposed)
  v16bf afr[12];
#pragma unroll
  for (int kt = 0; kt < 12; ++kt) afr[kt] = load_a_frag(bufA, wave * 16, MD, kt, lane);
#pragma unroll 1
  for (int ct = 0; ct < 48; ++ct) {
    v8f acc = {};
    acc = gemm_k12(afr, wkvs, ct, lane, acc);
    if (ct < 24) {
#pragma unroll
      for (int r = 0; r < 8; ++r) {
        size_t row = r0 + wave * 16 + h * 8 + r;
        kout[row * MD + ct * 16 + nn] = f2bf(acc[r]);
      }
    } else {
      size_t dcol = (size_t)((ct - 24) * 16 + nn);
#pragma unroll
      for (int r = 0; r < 8; ++r) {
        size_t row = r0 + wave * 16 + h * 8 + r;
        vTout[dcol * TOTROWS + row] = f2bf(acc[r]);
      }
    }
  }
}

// ---------------- q = LN(query) @ wq  (scale folded in) ----------------
__global__ void __launch_bounds__(256, 1)
qproj_kernel(const float* __restrict__ query,
             const float* __restrict__ lng, const float* __restrict__ lnb,
             const unsigned short* __restrict__ wqs,
             unsigned short* __restrict__ qout) {
  extern __shared__ char smem[];
  unsigned short* buf = (unsigned short*)smem;           // [128][384]
  float* stats = (float*)(buf + 128 * MD);
  int t = threadIdx.x, lane = t & 31, wave = t >> 5;
  int h = lane >> 4, nn = lane & 15;
  int r0 = blockIdx.x * 128;
  if (t < 128) {
    int row = r0 + t;
    float s = 0.f, ss = 0.f;
#pragma unroll 1
    for (int d = 0; d < MD; ++d) { float v = query[(size_t)row * MD + d]; s += v; ss += v * v; }
    float mean = s / MD, var = ss / MD - mean * mean;
    stats[2 * t] = mean; stats[2 * t + 1] = rsqrtf(var + 1e-5f);
  }
  __syncthreads();
  for (int e = t; e < 128 * MD; e += 256) {
    int rr = e / MD, d = e - rr * MD;
    float v = query[(size_t)(r0 + rr) * MD + d];
    buf[e] = f2bf((v - stats[2 * rr]) * stats[2 * rr + 1] * lng[d] + lnb[d]);
  }
  __syncthreads();
  v16bf afr[12];
#pragma unroll
  for (int kt = 0; kt < 12; ++kt) afr[kt] = load_a_frag(buf, wave * 16, MD, kt, lane);
  const float scale = 0.05103103630798288f;  // 384^-0.5 folded into q
#pragma unroll 1
  for (int ct = 0; ct < 24; ++ct) {
    v8f acc = {};
    acc = gemm_k12(afr, wqs, ct, lane, acc);
#pragma unroll
    for (int r = 0; r < 8; ++r) {
      int row = r0 + wave * 16 + h * 8 + r;
      qout[(size_t)row * MD + ct * 16 + nn] = f2bf(acc[r] * scale);
    }
  }
}

// ---------------- streaming (flash) cross-attention ----------------
// grid = 64 blocks x 256 threads; each wave owns one 16-row q tile.
__global__ void __launch_bounds__(256, 1)
attn_kernel(const unsigned short* __restrict__ q,
            const unsigned short* __restrict__ k,
            const unsigned short* __restrict__ vT,
            unsigned short* __restrict__ attnout) {
  extern __shared__ char smem[];
  int t = threadIdx.x, lane = t & 31, wave = t >> 5;
  unsigned short* pbuf = (unsigned short*)smem + wave * (16 * 32);  // per-wave p tile
  int h = lane >> 4, nn = lane & 15;
  int qt = blockIdx.x * 8 + wave;         // 0..511
  int b  = qt >> 5;                       // batch
  int ti = qt & 31;                       // q tile within batch
  size_t bn = (size_t)b * NPTS;
  size_t h16 = (size_t)(h * 16);

  v16bf qf[12];
#pragma unroll
  for (int kt = 0; kt < 12; ++kt) qf[kt] = load_a_frag(q, ti * 16, MD, kt, lane);

  v8f zero = {};
  v8f oacc[24];
#pragma unroll
  for (int c = 0; c < 24; ++c) oacc[c] = zero;
  float M[8], S[8];
#pragma unroll
  for (int r = 0; r < 8; ++r) { M[r] = -1e30f; S[r] = 0.f; }

#pragma unroll 1
  for (int chunk = 0; chunk < NPTS / 32; ++chunk) {
    int n0 = chunk * 32;
    // prefetch next chunk's k rows into cache (global_prefetch_b8)
    if (chunk + 1 < NPTS / 32) {
      const unsigned short* pf = k + (bn + n0 + 32 + nn) * (size_t)MD + h16;
      __builtin_prefetch(pf, 0, 0);
      __builtin_prefetch(pf + 16 * (size_t)MD, 0, 0);
    }
    // scores: two 16-col tiles, K = D = 384 (12 k-tiles), 2-deep pipeline
    v8f s0 = zero, s1 = zero;
    {
      size_t row0 = bn + n0 + nn;
      v16bf b0 = load_b_frag_rm(k, row0, MD, h16);
      v16bf b1 = load_b_frag_rm(k, row0, MD, 32 + h16);
#pragma unroll
      for (int kt = 0; kt < 12; kt += 2) {
        v16bf m0 = b0, m1 = b1;
        if (kt + 2 < 12) m0 = load_b_frag_rm(k, row0, MD, (size_t)((kt + 2) * 32) + h16);
        if (kt + 3 < 12) m1 = load_b_frag_rm(k, row0, MD, (size_t)((kt + 3) * 32) + h16);
        s0 = wmma_bf16(qf[kt],     b0, s0);
        s0 = wmma_bf16(qf[kt + 1], b1, s0);
        b0 = m0; b1 = m1;
      }
    }
    {
      size_t row1 = bn + n0 + 16 + nn;
      v16bf b0 = load_b_frag_rm(k, row1, MD, h16);
      v16bf b1 = load_b_frag_rm(k, row1, MD, 32 + h16);
#pragma unroll
      for (int kt = 0; kt < 12; kt += 2) {
        v16bf m0 = b0, m1 = b1;
        if (kt + 2 < 12) m0 = load_b_frag_rm(k, row1, MD, (size_t)((kt + 2) * 32) + h16);
        if (kt + 3 < 12) m1 = load_b_frag_rm(k, row1, MD, (size_t)((kt + 3) * 32) + h16);
        s1 = wmma_bf16(qf[kt],     b0, s1);
        s1 = wmma_bf16(qf[kt + 1], b1, s1);
        b0 = m0; b1 = m1;
      }
    }
    // online softmax (rows m = h*8 + r, cols striped over 16 lanes)
#pragma unroll
    for (int r = 0; r < 8; ++r) {
      float mx = fmaxf(s0[r], s1[r]);
#pragma unroll
      for (int mm = 1; mm < 16; mm <<= 1) mx = fmaxf(mx, __shfl_xor(mx, mm, 32));
      float mnew  = fmaxf(M[r], mx);
      float alpha = __expf(M[r] - mnew);
      float p0 = __expf(s0[r] - mnew);
      float p1 = __expf(s1[r] - mnew);
      float rs = p0 + p1;
#pragma unroll
      for (int mm = 1; mm < 16; mm <<= 1) rs += __shfl_xor(rs, mm, 32);
      S[r] = S[r] * alpha + rs;
      M[r] = mnew;
#pragma unroll
      for (int c = 0; c < 24; ++c) oacc[c][r] *= alpha;
      int mrow = h * 8 + r;
      pbuf[mrow * 32 + nn]      = f2bf(p0);
      pbuf[mrow * 32 + 16 + nn] = f2bf(p1);
    }
    asm volatile("s_wait_dscnt 0" ::: "memory");   // LDS p-tile visible to A-frag read
    v16bf pa = load_a_frag(pbuf, 0, 32, 0, lane);
    {
      v16bf b0 = load_b_frag_rm(vT, (size_t)nn,      TOTROWS, bn + n0 + h16);
      v16bf b1 = load_b_frag_rm(vT, (size_t)(16 + nn), TOTROWS, bn + n0 + h16);
#pragma unroll
      for (int c = 0; c < 24; c += 2) {
        v16bf m0 = b0, m1 = b1;
        if (c + 2 < 24)
          m0 = load_b_frag_rm(vT, (size_t)((c + 2) * 16 + nn), TOTROWS, bn + n0 + h16);
        if (c + 3 < 24)
          m1 = load_b_frag_rm(vT, (size_t)((c + 3) * 16 + nn), TOTROWS, bn + n0 + h16);
        oacc[c]     = wmma_bf16(pa, b0, oacc[c]);
        oacc[c + 1] = wmma_bf16(pa, b1, oacc[c + 1]);
        b0 = m0; b1 = m1;
      }
    }
  }
  // epilogue: divide by softmax denominator
  size_t qrow0 = (size_t)b * LQ + ti * 16;
#pragma unroll 1
  for (int c = 0; c < 24; ++c)
#pragma unroll
    for (int r = 0; r < 8; ++r) {
      size_t row = qrow0 + h * 8 + r;
      attnout[row * MD + c * 16 + nn] = f2bf(oacc[c][r] / S[r]);
    }
}

// ---------------- wo-proj + residual + LN + GEGLU FF + residual ----------------
// grid = 64 blocks (128 rows each) x 256 threads.
__global__ void __launch_bounds__(256, 1)
ffn_kernel(const unsigned short* __restrict__ attnout,
           const unsigned short* __restrict__ wos,
           const float* __restrict__ bo, const float* __restrict__ query,
           const float* __restrict__ lng, const float* __restrict__ lnb,
           const unsigned short* __restrict__ w1s, const float* __restrict__ fb1,
           const unsigned short* __restrict__ w2s, const float* __restrict__ fb2,
           float* __restrict__ out) {
  extern __shared__ char smem[];
  float* x1 = (float*)smem;                                     // [128][384] f32
  unsigned short* lnbuf = (unsigned short*)(x1 + 128 * MD);     // [128][384] bf16
  float* stats = (float*)(lnbuf + 128 * MD);                    // [128][2]
  unsigned short* pball = (unsigned short*)(stats + 256);       // 8 x [16][32]
  int t = threadIdx.x, lane = t & 31, wave = t >> 5;
  unsigned short* pbuf = pball + wave * (16 * 32);
  int h = lane >> 4, nn = lane & 15;
  size_t r0 = (size_t)blockIdx.x * 128;

  // x1 = attnout @ wo + bo + lq
  {
    v16bf afr[12];
#pragma unroll
    for (int kt = 0; kt < 12; ++kt)
      afr[kt] = load_a_frag(attnout, (int)r0 + wave * 16, MD, kt, lane);
#pragma unroll 1
    for (int ct = 0; ct < 24; ++ct) {
      v8f acc = {};
      acc = gemm_k12(afr, wos, ct, lane, acc);
      int n = ct * 16 + nn;
      float bias = bo[n];
#pragma unroll
      for (int r = 0; r < 8; ++r) {
        int mloc = wave * 16 + h * 8 + r;
        int li = (int)((r0 + mloc) % LQ);
        x1[mloc * MD + n] = acc[r] + bias + query[(size_t)li * MD + n];
      }
    }
  }
  __syncthreads();
  // LN(x1)
  if (t < 128) {
    float s = 0.f, ss = 0.f;
#pragma unroll 1
    for (int d = 0; d < MD; ++d) { float v = x1[t * MD + d]; s += v; ss += v * v; }
    float mean = s / MD, var = ss / MD - mean * mean;
    stats[2 * t] = mean; stats[2 * t + 1] = rsqrtf(var + 1e-5f);
  }
  __syncthreads();
  for (int e = t; e < 128 * MD; e += 256) {
    int rr = e / MD, d = e - rr * MD;
    lnbuf[e] = f2bf((x1[e] - stats[2 * rr]) * stats[2 * rr + 1] * lng[d] + lnb[d]);
  }
  __syncthreads();

  // GEGLU FF streamed over FFI in 32-col chunks, accumulate out in registers
  v16bf lfr[12];
#pragma unroll
  for (int kt = 0; kt < 12; ++kt) lfr[kt] = load_a_frag(lnbuf, wave * 16, MD, kt, lane);
  v8f zero = {};
  v8f oacc[24];
#pragma unroll
  for (int c = 0; c < 24; ++c) oacc[c] = zero;
  const float inv_sqrt2 = 0.70710678118654752f;
#pragma unroll 1
  for (int chunk = 0; chunk < FFI / 32; ++chunk) {
    v8f a0 = zero, a1 = zero, g0 = zero, g1 = zero;
    int cta = chunk * 2;
    int ctg = (FFI >> 4) + chunk * 2;   // gate half starts at col FFI
#pragma unroll
    for (int kt = 0; kt < 12; ++kt) {
      a0 = wmma_bf16(lfr[kt], load_b_frag_swz(w1s, cta,     kt, 12, lane), a0);
      a1 = wmma_bf16(lfr[kt], load_b_frag_swz(w1s, cta + 1, kt, 12, lane), a1);
      g0 = wmma_bf16(lfr[kt], load_b_frag_swz(w1s, ctg,     kt, 12, lane), g0);
      g1 = wmma_bf16(lfr[kt], load_b_frag_swz(w1s, ctg + 1, kt, 12, lane), g1);
    }
    int ca0 = chunk * 32 + nn, ca1 = chunk * 32 + 16 + nn;
    float ba0 = fb1[ca0], ba1 = fb1[ca1];
    float bg0 = fb1[FFI + ca0], bg1 = fb1[FFI + ca1];
#pragma unroll
    for (int r = 0; r < 8; ++r) {
      int mrow = h * 8 + r;
      float av0 = a0[r] + ba0, av1 = a1[r] + ba1;
      float gv0 = g0[r] + bg0, gv1 = g1[r] + bg1;
      float f0 = av0 * (0.5f * gv0 * (1.f + erff(gv0 * inv_sqrt2)));
      float f1 = av1 * (0.5f * gv1 * (1.f + erff(gv1 * inv_sqrt2)));
      pbuf[mrow * 32 + nn]      = f2bf(f0);
      pbuf[mrow * 32 + 16 + nn] = f2bf(f1);
    }
    asm volatile("s_wait_dscnt 0" ::: "memory");
    v16bf pa = load_a_frag(pbuf, 0, 32, 0, lane);
    {
      v16bf b0 = load_b_frag_swz(w2s, 0, chunk, 48, lane);
      v16bf b1 = load_b_frag_swz(w2s, 1, chunk, 48, lane);
#pragma unroll
      for (int c = 0; c < 24; c += 2) {
        v16bf m0 = b0, m1 = b1;
        if (c + 2 < 24) m0 = load_b_frag_swz(w2s, c + 2, chunk, 48, lane);
        if (c + 3 < 24) m1 = load_b_frag_swz(w2s, c + 3, chunk, 48, lane);
        oacc[c]     = wmma_bf16(pa, b0, oacc[c]);
        oacc[c + 1] = wmma_bf16(pa, b1, oacc[c + 1]);
        b0 = m0; b1 = m1;
      }
    }
  }
  // out = f @ ff_w2 + fb2 + x1
#pragma unroll 1
  for (int c = 0; c < 24; ++c) {
    int n = c * 16 + nn;
    float b2v = fb2[n];
#pragma unroll
    for (int r = 0; r < 8; ++r) {
      int mloc = wave * 16 + h * 8 + r;
      out[(r0 + mloc) * MD + n] = oacc[c][r] + b2v + x1[mloc * MD + n];
    }
  }
}

// ---------------- host launcher ----------------
extern "C" void kernel_launch(void* const* d_in, const int* in_sizes, int n_in,
                              void* d_out, int out_size, void* d_ws, size_t ws_size,
                              hipStream_t stream) {
  (void)in_sizes; (void)n_in; (void)out_size; (void)ws_size;
  const float* x     = (const float*)d_in[0];
  const float* mw0   = (const float*)d_in[1];
  const float* mb0   = (const float*)d_in[2];
  const float* mw1   = (const float*)d_in[3];
  const float* mb1   = (const float*)d_in[4];
  const float* mw2   = (const float*)d_in[5];
  const float* mb2   = (const float*)d_in[6];
  const float* mw3   = (const float*)d_in[7];
  const float* mb3   = (const float*)d_in[8];
  const float* query = (const float*)d_in[9];
  const float* lnqg  = (const float*)d_in[10];
  const float* lnqb  = (const float*)d_in[11];
  const float* lncg  = (const float*)d_in[12];
  const float* lncb  = (const float*)d_in[13];
  const float* wq    = (const float*)d_in[14];
  const float* wkv   = (const float*)d_in[15];
  const float* wo    = (const float*)d_in[16];
  const float* bo    = (const float*)d_in[17];
  const float* lnfg  = (const float*)d_in[18];
  const float* lnfb  = (const float*)d_in[19];
  const float* fw1   = (const float*)d_in[20];
  const float* fb1   = (const float*)d_in[21];
  const float* fw2   = (const float*)d_in[22];
  const float* fb2   = (const float*)d_in[23];
  float* out = (float*)d_out;

  char* ws = (char*)d_ws;
  size_t off = 0;
  auto alloc = [&](size_t bytes) -> char* {
    char* p = ws + off;
    off += (bytes + 255) & ~(size_t)255;
    return p;
  };
  unsigned short* w1s   = (unsigned short*)alloc((size_t)MD * MD * 2);
  unsigned short* w2s   = (unsigned short*)alloc((size_t)MD * MD * 2);
  unsigned short* w3s   = (unsigned short*)alloc((size_t)MD * MD * 2);
  unsigned short* wqs   = (unsigned short*)alloc((size_t)MD * MD * 2);
  unsigned short* wkvs  = (unsigned short*)alloc((size_t)MD * 2 * MD * 2);
  unsigned short* wos   = (unsigned short*)alloc((size_t)MD * MD * 2);
  unsigned short* fw1s  = (unsigned short*)alloc((size_t)MD * 2 * FFI * 2);
  unsigned short* fw2s  = (unsigned short*)alloc((size_t)FFI * MD * 2);
  unsigned short* qbuf  = (unsigned short*)alloc((size_t)LQ * MD * 2);
  unsigned short* abuf  = (unsigned short*)alloc((size_t)BATCH * LQ * MD * 2);
  unsigned short* kbuf  = (unsigned short*)alloc((size_t)TOTROWS * MD * 2);
  unsigned short* vTbuf = (unsigned short*)alloc((size_t)TOTROWS * MD * 2);

  auto swz = [&](const float* src, unsigned short* dst, int K, int N) {
    int total = K * N;
    int grid = (total + 255) / 256;
    swz_kernel<<<grid, 256, 0, stream>>>(src, dst, K, N);
  };
  swz(mw1, w1s, MD, MD);
  swz(mw2, w2s, MD, MD);
  swz(mw3, w3s, MD, MD);
  swz(wq,  wqs, MD, MD);
  swz(wkv, wkvs, MD, 2 * MD);
  swz(wo,  wos, MD, MD);
  swz(fw1, fw1s, MD, 2 * FFI);
  swz(fw2, fw2s, FFI, MD);

  // fused point MLP + LN + kv
  {
    size_t smem = (size_t)2 * 128 * MD * 2 + 128 * 2 * 4;   // 197632 B
    mlp_kernel<<<TOTROWS / 128, 256, smem, stream>>>(
        x, mw0, mb0, w1s, mb1, w2s, mb2, w3s, mb3, lncg, lncb, wkvs, kbuf, vTbuf);
  }
  // q projection (shared across batch)
  {
    size_t smem = (size_t)128 * MD * 2 + 128 * 2 * 4;       // 99328 B
    qproj_kernel<<<LQ / 128, 256, smem, stream>>>(query, lnqg, lnqb, wqs, qbuf);
  }
  // streaming attention
  {
    size_t smem = (size_t)8 * 16 * 32 * 2;                  // 8192 B
    attn_kernel<<<(BATCH * (LQ / 16)) / 8, 256, smem, stream>>>(qbuf, kbuf, vTbuf, abuf);
  }
  // wo-proj + residual + LN + GEGLU FF
  {
    size_t smem = (size_t)128 * MD * 4 + (size_t)128 * MD * 2 + 256 * 4
                + (size_t)8 * 16 * 32 * 2;                  // 304128 B
    ffn_kernel<<<(BATCH * LQ) / 128, 256, smem, stream>>>(
        abuf, wos, bo, query, lnfg, lnfb, fw1s, fb1, fw2s, fb2, out);
  }
}
